// GraphTransformerLayer_996432413195
// MI455X (gfx1250) — compile-verified, matrix operand order checked by reading, not run on hardware
//
#include <hip/hip_runtime.h>
#include <hip/hip_bf16.h>

// ---------------------------------------------------------------------------
// GraphTransformerLayer for MI455X (gfx1250, wave32, WMMA)
//   N=100000 nodes, E=1.6M edges, DIM=64, H=4, HD=16, FF=256
// Weights are pre-swizzled into per-lane WMMA B-fragment register images so
// each v_wmma operand load is 2x global_load_b128 from L2-resident memory.
// ---------------------------------------------------------------------------

typedef __attribute__((ext_vector_type(16))) _Float16 v16h;
typedef __attribute__((ext_vector_type(8)))  _Float16 v8h;
typedef __attribute__((ext_vector_type(8)))  float    v8f;

#define NDIM 64
#define NH   4
#define HD   16
#define NFF  256

// ---------------- WMMA fragment helpers (CDNA5 ISA 7.12.2 layouts) ---------

// 16-bit A-matrix 16x32 (MxK) from LDS, row-major with `stride` halves/row.
// Per lane the two 8-half element groups are contiguous -> 2x ds_load_b128.
__device__ __forceinline__ v16h load_a_lds(const _Float16* base, int stride,
                                           int k0, int lane) {
  int row   = lane & 15;
  int khalf = (lane >> 4) << 3;   // 0 or 8
  v16h a;
#pragma unroll
  for (int i = 0; i < 8; ++i) {
    int kk = ((i < 4) ? (2 * i) : (16 + 2 * (i - 4))) + khalf + k0;
    a[2 * i]     = base[row * stride + kk];
    a[2 * i + 1] = base[row * stride + kk + 1];
  }
  return a;
}

// Pre-swizzled B fragment: fragment (kt,nt) stored as [32 lanes][16 halves].
__device__ __forceinline__ v16h load_b_pk(const _Float16* __restrict__ packed,
                                          int nkt, int kt, int nt, int lane) {
  return *(const v16h*)(packed + (((size_t)nt * nkt + kt) * 32 + lane) * 16);
}

__device__ __forceinline__ v8f wmma_f16(v16h a, v16h b, v8f c) {
  return __builtin_amdgcn_wmma_f32_16x16x32_f16(
      /*neg_a=*/false, a, /*neg_b=*/false, b,
      /*c_mod=*/(short)0, c, /*reuse_a=*/false, /*reuse_b=*/false);
}

// ---------------- utility kernels ------------------------------------------

// Swizzle a row-major f32 weight W[Kdim][ncols] into f16 WMMA B-fragment
// register images: dst[((nt*nkt+kt)*32 + lane)*16 + j].
__global__ void k_pack_b(const float* __restrict__ W, _Float16* __restrict__ dst,
                         int ncols, int nkt, int nnt) {
  int t = blockIdx.x * blockDim.x + threadIdx.x;
  int total = nkt * nnt * 32;
  if (t >= total) return;
  int lane = t & 31;
  int frag = t >> 5;
  int kt = frag % nkt;
  int nt = frag / nkt;
  int col   = (lane & 15) + nt * 16;
  int khalf = (lane >> 4) << 3;
#pragma unroll
  for (int j = 0; j < 16; ++j) {
    int i  = j >> 1;
    int kk = ((i < 4) ? (2 * i) : (16 + 2 * (i - 4))) + khalf + kt * 32 + (j & 1);
    dst[((size_t)frag * 32 + lane) * 16 + j] = (_Float16)W[(size_t)kk * ncols + col];
  }
}

__global__ void k_zero(float* __restrict__ p, int n) {
  int i = blockIdx.x * blockDim.x + threadIdx.x;
  if (i < n) p[i] = 0.0f;
}

// ---------------- kernel 1: LN1 + Q/K/V projections (WMMA) -----------------
// One wave per 16-node tile; 4 waves per block.

__global__ void __launch_bounds__(128)
k_ln_qkv(const float* __restrict__ x,
         const float* __restrict__ g1, const float* __restrict__ b1,
         const _Float16* __restrict__ wq_p, const _Float16* __restrict__ wk_p,
         const _Float16* __restrict__ wv_p,
         float* __restrict__ Q, float* __restrict__ K, float* __restrict__ V,
         int ntiles) {
  __shared__ _Float16 s_xn[4][16 * 72];   // row pitch 144B: banked + 16B aligned
  int lane = threadIdx.x & 31;
  int wid  = threadIdx.x >> 5;
  int tile = blockIdx.x * 4 + wid;
  bool active = (tile < ntiles);          // wave-uniform
  int nbase = tile * 16;

  if (active && lane < 16) {
    const float4* xr4 = (const float4*)(x + (size_t)(nbase + lane) * NDIM);
    float4 xv[16];
    float s = 0.f, ss = 0.f;
#pragma unroll
    for (int q = 0; q < 16; ++q) {
      xv[q] = xr4[q];
      s  += xv[q].x + xv[q].y + xv[q].z + xv[q].w;
      ss += xv[q].x * xv[q].x + xv[q].y * xv[q].y +
            xv[q].z * xv[q].z + xv[q].w * xv[q].w;
    }
    float m   = s * (1.0f / NDIM);
    float var = ss * (1.0f / NDIM) - m * m;
    var = var > 0.f ? var : 0.f;
    float inv = rsqrtf(var + 1e-5f);
    const float* xs = (const float*)xv;
#pragma unroll
    for (int c8 = 0; c8 < 8; ++c8) {
      v8h t;
#pragma unroll
      for (int k = 0; k < 8; ++k) {
        int c = c8 * 8 + k;
        t[k] = (_Float16)((xs[c] - m) * inv * g1[c] + b1[c]);
      }
      *(v8h*)(&s_xn[wid][lane * 72 + c8 * 8]) = t;   // ds_store_b128
    }
  }
  __syncthreads();
  if (!active) return;

  v16h a0 = load_a_lds(s_xn[wid], 72, 0, lane);
  v16h a1 = load_a_lds(s_xn[wid], 72, 32, lane);

  const _Float16* Ws[3] = {wq_p, wk_p, wv_p};
  float* Os[3]          = {Q, K, V};
#pragma unroll
  for (int w = 0; w < 3; ++w) {
#pragma unroll
    for (int n = 0; n < 4; ++n) {
      v8f acc = {};
      acc = wmma_f16(a0, load_b_pk(Ws[w], 2, 0, n, lane), acc);
      acc = wmma_f16(a1, load_b_pk(Ws[w], 2, 1, n, lane), acc);
      int col   = (lane & 15) + n * 16;
      int rbase = (lane >> 4) << 3;
      float* o  = Os[w] + (size_t)nbase * NDIM;
#pragma unroll
      for (int v = 0; v < 8; ++v) o[(rbase + v) * NDIM + col] = acc[v];
    }
  }
}

// ---------------- kernel 2: per-edge attention, fused scatter --------------
// alpha = <Q[row,h],K[col,h]>/4 ; scatter exp(alpha) into denom[row,h] and
// exp(alpha)*V[col] into outU[row]  (normalization deferred to kernel 3).

__global__ void __launch_bounds__(256)
k_edge(const int* __restrict__ rows, const int* __restrict__ cols,
       const float* __restrict__ Q, const float* __restrict__ Km,
       const float* __restrict__ Vm,
       float* __restrict__ denom, float* __restrict__ outU, int nedges) {
  int e = blockIdx.x * blockDim.x + threadIdx.x;
  if (e >= nedges) return;
  int r = rows[e];
  int c = cols[e];
  const float4* q4 = (const float4*)(Q  + (size_t)r * NDIM);
  const float4* k4 = (const float4*)(Km + (size_t)c * NDIM);
  const float4* v4 = (const float4*)(Vm + (size_t)c * NDIM);

  float aexp[NH];
#pragma unroll
  for (int h = 0; h < NH; ++h) {
    float d = 0.f;
#pragma unroll
    for (int i = 0; i < 4; ++i) {
      float4 q = q4[h * 4 + i], k = k4[h * 4 + i];
      d += q.x * k.x + q.y * k.y + q.z * k.z + q.w * k.w;
    }
    float ae = __expf(d * 0.25f);       // 1/sqrt(HD) = 0.25
    aexp[h] = ae;
    atomicAdd(denom + (size_t)r * NH + h, ae);
  }
  float* o = outU + (size_t)r * NDIM;
#pragma unroll
  for (int h = 0; h < NH; ++h) {
#pragma unroll
    for (int i = 0; i < 4; ++i) {
      float4 v = v4[h * 4 + i];
      float a = aexp[h];
      atomicAdd(o + h * HD + i * 4 + 0, v.x * a);
      atomicAdd(o + h * HD + i * 4 + 1, v.y * a);
      atomicAdd(o + h * HD + i * 4 + 2, v.z * a);
      atomicAdd(o + h * HD + i * 4 + 3, v.w * a);
    }
  }
}

// ---------------- kernel 3: normalize, @w_o, residual (WMMA) ---------------

__global__ void __launch_bounds__(128)
k_attnout(const float* __restrict__ x, const float* __restrict__ outU,
          const float* __restrict__ denom, const _Float16* __restrict__ wo_p,
          float* __restrict__ x1, int ntiles) {
  __shared__ _Float16 s_a[4][16 * 72];
  int lane = threadIdx.x & 31;
  int wid  = threadIdx.x >> 5;
  int tile = blockIdx.x * 4 + wid;
  bool active = (tile < ntiles);
  int nbase = tile * 16;

  if (active && lane < 16) {
    int node = nbase + lane;
    float dn[NH];
#pragma unroll
    for (int h = 0; h < NH; ++h)
      dn[h] = 1.0f / (denom[(size_t)node * NH + h] + 1e-16f);
    const float4* o4 = (const float4*)(outU + (size_t)node * NDIM);
#pragma unroll
    for (int c8 = 0; c8 < 8; ++c8) {
      float4 a = o4[c8 * 2], b = o4[c8 * 2 + 1];
      float r = dn[c8 >> 1];        // 16 consecutive cols per head
      v8h t;
      t[0] = (_Float16)(a.x * r); t[1] = (_Float16)(a.y * r);
      t[2] = (_Float16)(a.z * r); t[3] = (_Float16)(a.w * r);
      t[4] = (_Float16)(b.x * r); t[5] = (_Float16)(b.y * r);
      t[6] = (_Float16)(b.z * r); t[7] = (_Float16)(b.w * r);
      *(v8h*)(&s_a[wid][lane * 72 + c8 * 8]) = t;
    }
  }
  __syncthreads();
  if (!active) return;

  v16h a0 = load_a_lds(s_a[wid], 72, 0, lane);
  v16h a1 = load_a_lds(s_a[wid], 72, 32, lane);
#pragma unroll
  for (int n = 0; n < 4; ++n) {
    v8f acc = {};
    acc = wmma_f16(a0, load_b_pk(wo_p, 2, 0, n, lane), acc);
    acc = wmma_f16(a1, load_b_pk(wo_p, 2, 1, n, lane), acc);
    int col   = (lane & 15) + n * 16;
    int rbase = (lane >> 4) << 3;
#pragma unroll
    for (int v = 0; v < 8; ++v) {
      size_t idx = (size_t)(nbase + rbase + v) * NDIM + col;
      x1[idx] = x[idx] + acc[v];
    }
  }
}

// ---------------- kernel 4: LN2 + FFN (64->256 relu 256->64) + residual ----

__global__ void __launch_bounds__(128)
k_ffn(const float* __restrict__ x1,
      const float* __restrict__ g2, const float* __restrict__ b2,
      const _Float16* __restrict__ f1_p, const float* __restrict__ f1b,
      const _Float16* __restrict__ f2_p, const float* __restrict__ f2b,
      float* __restrict__ out, int ntiles) {
  __shared__ _Float16 s_xn[4][16 * 72];
  __shared__ _Float16 s_h[4][16 * 264];   // [16,256] hidden tile, padded
  int lane = threadIdx.x & 31;
  int wid  = threadIdx.x >> 5;
  int tile = blockIdx.x * 4 + wid;
  bool active = (tile < ntiles);
  int nbase = tile * 16;

  if (active && lane < 16) {
    const float4* xr4 = (const float4*)(x1 + (size_t)(nbase + lane) * NDIM);
    float4 xv[16];
    float s = 0.f, ss = 0.f;
#pragma unroll
    for (int q = 0; q < 16; ++q) {
      xv[q] = xr4[q];
      s  += xv[q].x + xv[q].y + xv[q].z + xv[q].w;
      ss += xv[q].x * xv[q].x + xv[q].y * xv[q].y +
            xv[q].z * xv[q].z + xv[q].w * xv[q].w;
    }
    float m   = s * (1.0f / NDIM);
    float var = ss * (1.0f / NDIM) - m * m;
    var = var > 0.f ? var : 0.f;
    float inv = rsqrtf(var + 1e-5f);
    const float* xs = (const float*)xv;
#pragma unroll
    for (int c8 = 0; c8 < 8; ++c8) {
      v8h t;
#pragma unroll
      for (int k = 0; k < 8; ++k) {
        int c = c8 * 8 + k;
        t[k] = (_Float16)((xs[c] - m) * inv * g2[c] + b2[c]);
      }
      *(v8h*)(&s_xn[wid][lane * 72 + c8 * 8]) = t;
    }
  }
  __syncthreads();

  if (active) {
    v16h a0 = load_a_lds(s_xn[wid], 72, 0, lane);
    v16h a1 = load_a_lds(s_xn[wid], 72, 32, lane);
#pragma unroll
    for (int n = 0; n < 16; ++n) {          // 256 output cols
      v8f acc = {};
      acc = wmma_f16(a0, load_b_pk(f1_p, 2, 0, n, lane), acc);
      acc = wmma_f16(a1, load_b_pk(f1_p, 2, 1, n, lane), acc);
      int col   = (lane & 15) + n * 16;
      int rbase = (lane >> 4) << 3;
      float bias = f1b[col];
#pragma unroll
      for (int v = 0; v < 8; ++v) {
        float t = acc[v] + bias;
        t = t > 0.f ? t : 0.f;               // relu
        s_h[wid][(rbase + v) * 264 + col] = (_Float16)t;
      }
    }
  }
  __syncthreads();

  if (active) {
#pragma unroll
    for (int n = 0; n < 4; ++n) {
      v8f acc = {};
#pragma unroll
      for (int kt = 0; kt < 8; ++kt) {       // K = 256
        v16h a = load_a_lds(s_h[wid], 264, kt * 32, lane);
        v16h b = load_b_pk(f2_p, 8, kt, n, lane);
        acc = wmma_f16(a, b, acc);
      }
      int col   = (lane & 15) + n * 16;
      int rbase = (lane >> 4) << 3;
      float bias = f2b[col];
#pragma unroll
      for (int v = 0; v < 8; ++v) {
        size_t idx = (size_t)(nbase + rbase + v) * NDIM + col;
        out[idx] = x1[idx] + acc[v] + bias;
      }
    }
  }
}

// ---------------- host orchestration ---------------------------------------

extern "C" void kernel_launch(void* const* d_in, const int* in_sizes, int n_in,
                              void* d_out, int out_size, void* d_ws, size_t ws_size,
                              hipStream_t stream) {
  const float* x     = (const float*)d_in[0];
  const int*   ei    = (const int*)  d_in[1];
  const float* w_q   = (const float*)d_in[2];
  const float* w_k   = (const float*)d_in[3];
  const float* w_v   = (const float*)d_in[4];
  const float* w_o   = (const float*)d_in[5];
  const float* ln1_g = (const float*)d_in[6];
  const float* ln1_b = (const float*)d_in[7];
  const float* ln2_g = (const float*)d_in[8];
  const float* ln2_b = (const float*)d_in[9];
  const float* ff1_w = (const float*)d_in[10];
  const float* ff1_b = (const float*)d_in[11];
  const float* ff2_w = (const float*)d_in[12];
  const float* ff2_b = (const float*)d_in[13];
  float* out = (float*)d_out;

  const int N = in_sizes[0] / NDIM;       // 100000
  const int E = in_sizes[1] / 2;          // 1600000
  const int ntiles = (N + 15) / 16;       // N divisible by 16 here

  // workspace layout (packed B-fragment weight images, then f32 tensors)
  char* ws = (char*)d_ws;
  _Float16* wq_p = (_Float16*)(ws + 0);        // 4096 halves
  _Float16* wk_p = (_Float16*)(ws + 8192);
  _Float16* wv_p = (_Float16*)(ws + 16384);
  _Float16* wo_p = (_Float16*)(ws + 24576);
  _Float16* f1_p = (_Float16*)(ws + 32768);    // 16384 halves
  _Float16* f2_p = (_Float16*)(ws + 65536);    // 16384 halves
  float* Q     = (float*)(ws + 98304);
  float* K     = Q + (size_t)N * NDIM;
  float* V     = K + (size_t)N * NDIM;
  float* denom = V + (size_t)N * NDIM;         // [N,4]
  float* outU  = denom + (size_t)N * NH;       // [N,64] (contiguous after denom)
  float* x1    = outU + (size_t)N * NDIM;

  const int* rows = ei;
  const int* cols = ei + E;

  // 1) swizzle weights into WMMA B-fragment images (f16)
  k_pack_b<<<1, 256, 0, stream>>>(w_q, wq_p, NDIM, 2, 4);
  k_pack_b<<<1, 256, 0, stream>>>(w_k, wk_p, NDIM, 2, 4);
  k_pack_b<<<1, 256, 0, stream>>>(w_v, wv_p, NDIM, 2, 4);
  k_pack_b<<<1, 256, 0, stream>>>(w_o, wo_p, NDIM, 2, 4);
  k_pack_b<<<4, 256, 0, stream>>>(ff1_w, f1_p, NFF, 2, 16);
  k_pack_b<<<4, 256, 0, stream>>>(ff2_w, f2_p, NDIM, 8, 4);

  // 2) zero accumulators (denom + outU are contiguous)
  int nz = N * NH + N * NDIM;
  k_zero<<<(nz + 255) / 256, 256, 0, stream>>>(denom, nz);

  // 3) LN1 + QKV projections
  k_ln_qkv<<<(ntiles + 3) / 4, 128, 0, stream>>>(x, ln1_g, ln1_b,
                                                 wq_p, wk_p, wv_p, Q, K, V, ntiles);
  // 4) edge attention + fused unnormalized scatter
  k_edge<<<(E + 255) / 256, 256, 0, stream>>>(rows, cols, Q, K, V, denom, outU, E);

  // 5) normalize + @w_o + residual
  k_attnout<<<(ntiles + 3) / 4, 128, 0, stream>>>(x, outU, denom, wo_p, x1, ntiles);

  // 6) LN2 + FFN + residual -> out
  k_ffn<<<(ntiles + 3) / 4, 128, 0, stream>>>(x1, ln2_g, ln2_b, f1_p, ff1_b,
                                              f2_p, ff2_b, out, ntiles);
}
